// TransformerBlock_81312320848142
// MI455X (gfx1250) — matrix operand order repaired and checked by reading, not compiled, not run on hardware
//
#include <hip/hip_runtime.h>
#include <hip/hip_bf16.h>
#include <math.h>

// ---------------------------------------------------------------------------
// CDNA5 (gfx1250) causal multi-head attention block, bf16 WMMA pipeline.
// GEMM staging via Tensor Data Mover (tensor_load_to_lds + s_wait_tensorcnt),
// double-buffered in LDS.
// ---------------------------------------------------------------------------

typedef __attribute__((ext_vector_type(16))) __bf16 v16bf;
typedef __attribute__((ext_vector_type(8)))  float  v8f;
typedef __attribute__((ext_vector_type(4)))  unsigned int v4u;
typedef __attribute__((ext_vector_type(8)))  int    v8i;
typedef __attribute__((ext_vector_type(4)))  int    v4i;

#if defined(__has_builtin)
#if __has_builtin(__builtin_amdgcn_tensor_load_to_lds) && \
    __has_builtin(__builtin_amdgcn_s_wait_tensorcnt)
#define HAVE_TDM 1
#endif
#endif
#ifndef HAVE_TDM
#define HAVE_TDM 0
#endif

__device__ __forceinline__ v8f wmma_bf16(v16bf a, v16bf b, v8f c) {
  // v_wmma_f32_16x16x32_bf16
  return __builtin_amdgcn_wmma_f32_16x16x32_bf16(
      /*neg_a=*/false, a, /*neg_b=*/false, b,
      /*c_mod=*/(short)0, c, /*reuse_a=*/false, /*reuse_b=*/false);
}

union Frag16 { v16bf v; uint4 q[2]; };

// A-matrix fragment (16x32 bf16). Row (M = lane&15) is contiguous with
// `stride` elements. Per ISA 7.12.2: lanes 0-15 hold K=0..7 then K=16..23,
// lanes 16-31 hold K=8..15 then K=24..31.
__device__ __forceinline__ v16bf load_frag_a(const __bf16* base, size_t stride, int lane) {
  const __bf16* p = base + (size_t)(lane & 15) * stride + (size_t)(((lane >> 4) & 1) * 8);
  Frag16 f;
  f.q[0] = *(const uint4*)(p);
  f.q[1] = *(const uint4*)(p + 16);
  return f.v;
}

// B-matrix fragment (32x16 bf16) where the N-indexed row is contiguous along
// K (exactly the torch [out,in] weight layout). lane&15 = N; lanes 0-15 hold
// K=0..7 then K=16..23, lanes 16-31 hold K=8..15 then K=24..31, packed 2 per VGPR.
__device__ __forceinline__ v16bf load_frag_b(const __bf16* base, size_t stride, int lane) {
  const __bf16* p = base + (size_t)(lane & 15) * stride + (size_t)(((lane >> 4) & 1) * 8);
  Frag16 f;
  f.q[0] = *(const uint4*)(p);
  f.q[1] = *(const uint4*)(p + 16);
  return f.v;
}

// ---------------------------------------------------------------------------
// TDM: issue a 2-D tensor tile load (bf16 elements) into LDS.
// D# layout per CDNA5 ISA ch.8:
//   group0: [1:0]=count=1, [63:32]=lds_addr, [120:64]=global_addr, [127:126]=2
//   group1: [17:16]=data_size(1 => 2B), [79:48]=tensor_dim0, [111:80]=tensor_dim1,
//           [127:112]=tile_dim0, [143:128]=tile_dim1, [207:160]=tensor_dim0_stride
// ---------------------------------------------------------------------------
__device__ __forceinline__ void tdm_load_tile_2d(
    const __bf16* gptr, unsigned int lds_off,
    unsigned int tile_d0, unsigned int tile_d1,
    unsigned int tensor_d0, unsigned int tensor_d1,
    unsigned int stride0_elems) {
#if HAVE_TDM
  unsigned long long ga = (unsigned long long)(uintptr_t)gptr;
  v4u g0 = { 1u,                                     // count = 1 valid descriptor
             lds_off,                                // LDS destination (bytes)
             (unsigned int)ga,                       // global_addr[31:0]
             ((unsigned int)(ga >> 32) & 0x01FFFFFFu) | 0x80000000u };  // addr[56:32] | type=2
  v8i g1 = {
      (int)(1u << 16),                                           // data_size = 2 bytes
      (int)((tensor_d0 & 0xFFFFu) << 16),                        // tensor_dim0[15:0]
      (int)((tensor_d0 >> 16) | ((tensor_d1 & 0xFFFFu) << 16)),  // dim0 hi | dim1 lo
      (int)((tensor_d1 >> 16) | (tile_d0 << 16)),                // dim1 hi | tile_dim0
      (int)(tile_d1 & 0xFFFFu),                                  // tile_dim1 (tile_dim2=0)
      (int)stride0_elems,                                        // tensor_dim0_stride[31:0]
      0, 0                                                       // stride hi / dim1_stride = 0
  };
  v4i gz = { 0, 0, 0, 0 };
#if __clang_major__ >= 23
  v8i gz8 = { 0, 0, 0, 0, 0, 0, 0, 0 };
  __builtin_amdgcn_tensor_load_to_lds(g0, g1, gz, gz, gz8, 0);
#else
  __builtin_amdgcn_tensor_load_to_lds(g0, g1, gz, gz, 0);
#endif
#else
  (void)gptr; (void)lds_off; (void)tile_d0; (void)tile_d1;
  (void)tensor_d0; (void)tensor_d1; (void)stride0_elems;
#endif
}

__device__ __forceinline__ unsigned int lds_addr_of(const void* p) {
  // Generic LDS pointers are {SHARED_BASE[63:32], offset[31:0]}: low 32 bits
  // are the byte offset the TDM descriptor wants.
  return (unsigned int)(uintptr_t)p;
}

// ---------------------------------------------------------------------------
// Elementwise f32 -> bf16 convert.
// ---------------------------------------------------------------------------
__global__ void f32_to_bf16(const float* __restrict__ in, __bf16* __restrict__ out, int n) {
  int i = blockIdx.x * blockDim.x + threadIdx.x;
  if (i < n) out[i] = (__bf16)in[i];
}

// ---------------------------------------------------------------------------
// V [B*S, D] -> Vt [B, H, 64, S]  (so P*V B-fragments are contiguous in key)
// ---------------------------------------------------------------------------
__global__ void transpose_v(const __bf16* __restrict__ V, __bf16* __restrict__ Vt,
                            int B_, int S, int D, int H) {
  int o = blockIdx.x * blockDim.x + threadIdx.x;
  int total = B_ * S * D;
  if (o >= total) return;
  int s = o % S;
  int t = o / S;
  int d = t % 64;
  t /= 64;
  int h = t % H;
  int b = t / H;
  Vt[o] = V[((size_t)(b * S + s)) * D + h * 64 + d];
}

// ---------------------------------------------------------------------------
// GEMM: C[M,N] = A[M,K] * W[N,K]^T   (torch Linear, bf16 in, f32 accum)
// Block tile 128x128, BK=32, 8 waves (4x2), wave tile 32x64 (2x4 WMMA tiles).
// A/B tiles staged by TDM into double-buffered LDS; wave 0 drives the DMA and
// TENSORcnt, the workgroup barrier publishes the landed tiles.
// ---------------------------------------------------------------------------
#define BM 128
#define BN 128
#define BK 32

template <typename OutT>
__global__ __launch_bounds__(256) void gemm_bf16_wmma(
    const __bf16* __restrict__ A,  // [M,K] row-major
    const __bf16* __restrict__ W,  // [N,K] row-major
    OutT* __restrict__ C,          // [M,N] row-major
    int M, int N, int K) {
  __shared__ __align__(16) __bf16 As[2][BM][BK];
  __shared__ __align__(16) __bf16 Bs[2][BN][BK];
  const int tid  = threadIdx.x;
  const int lane = tid & 31;
  const int wave = tid >> 5;
  const int bm = blockIdx.y * BM;
  const int bn = blockIdx.x * BN;
  const int wr = (wave >> 1) * 32;  // wave row offset inside block tile
  const int wc = (wave & 1) * 64;   // wave col offset inside block tile

  v8f acc[2][4];
#pragma unroll
  for (int i = 0; i < 2; ++i)
#pragma unroll
    for (int j = 0; j < 4; ++j) acc[i][j] = {};

#if HAVE_TDM
  const unsigned int aoff[2] = { lds_addr_of(&As[0][0][0]), lds_addr_of(&As[1][0][0]) };
  const unsigned int boff[2] = { lds_addr_of(&Bs[0][0][0]), lds_addr_of(&Bs[1][0][0]) };
  // Prologue: DMA first K-slab into buffer 0.
  if (tid < 32) {
    tdm_load_tile_2d(A + (size_t)bm * K, aoff[0], BK, BM, (unsigned)K, (unsigned)(M - bm), (unsigned)K);
    tdm_load_tile_2d(W + (size_t)bn * K, boff[0], BK, BN, (unsigned)K, (unsigned)(N - bn), (unsigned)K);
  }
  int buf = 0;
  for (int kk = 0; kk < K; kk += BK) {
    if (tid < 32) __builtin_amdgcn_s_wait_tensorcnt(0);  // current slab landed
    __syncthreads();
    if (tid < 32 && (kk + BK) < K) {                     // DMA next slab while computing
      tdm_load_tile_2d(A + (size_t)bm * K + kk + BK, aoff[buf ^ 1], BK, BM, (unsigned)K, (unsigned)(M - bm), (unsigned)K);
      tdm_load_tile_2d(W + (size_t)bn * K + kk + BK, boff[buf ^ 1], BK, BN, (unsigned)K, (unsigned)(N - bn), (unsigned)K);
    }
    v16bf af[2], bfm[4];
#pragma unroll
    for (int i = 0; i < 2; ++i) af[i] = load_frag_a(&As[buf][wr + i * 16][0], BK, lane);
#pragma unroll
    for (int j = 0; j < 4; ++j) bfm[j] = load_frag_b(&Bs[buf][wc + j * 16][0], BK, lane);
#pragma unroll
    for (int i = 0; i < 2; ++i)
#pragma unroll
      for (int j = 0; j < 4; ++j)
        acc[i][j] = wmma_bf16(af[i], bfm[j], acc[i][j]);
    __syncthreads();
    buf ^= 1;
  }
#else
  // Fallback: manual VGPR staging (single live buffer, index 0).
  const int lr = tid >> 1;         // staging row (0..127)
  const int ls = (tid & 1) * 16;   // staging K segment
  for (int kk = 0; kk < K; kk += BK) {
    const uint4* ga = (const uint4*)(A + (size_t)(bm + lr) * K + kk + ls);
    const uint4* gb = (const uint4*)(W + (size_t)(bn + lr) * K + kk + ls);
    if (kk + BK < K) {  // global_prefetch_b8 for next K slab
      __builtin_prefetch(A + (size_t)(bm + lr) * K + kk + BK + ls, 0, 1);
      __builtin_prefetch(W + (size_t)(bn + lr) * K + kk + BK + ls, 0, 1);
    }
    *(uint4*)&As[0][lr][ls]     = ga[0];
    *(uint4*)&As[0][lr][ls + 8] = ga[1];
    *(uint4*)&Bs[0][lr][ls]     = gb[0];
    *(uint4*)&Bs[0][lr][ls + 8] = gb[1];
    __syncthreads();
    v16bf af[2], bfm[4];
#pragma unroll
    for (int i = 0; i < 2; ++i) af[i] = load_frag_a(&As[0][wr + i * 16][0], BK, lane);
#pragma unroll
    for (int j = 0; j < 4; ++j) bfm[j] = load_frag_b(&Bs[0][wc + j * 16][0], BK, lane);
#pragma unroll
    for (int i = 0; i < 2; ++i)
#pragma unroll
      for (int j = 0; j < 4; ++j)
        acc[i][j] = wmma_bf16(af[i], bfm[j], acc[i][j]);
    __syncthreads();
  }
#endif

  // C/D layout: lane&15 = N col; VGPR r holds M = r (lanes 0-15), r+8 (16-31)
  const int mofs = (lane >> 4) ? 8 : 0;
  const int nlan = lane & 15;
#pragma unroll
  for (int i = 0; i < 2; ++i)
#pragma unroll
    for (int j = 0; j < 4; ++j)
#pragma unroll
      for (int r = 0; r < 8; ++r) {
        int row = bm + wr + i * 16 + r + mofs;
        int col = bn + wc + j * 16 + nlan;
        C[(size_t)row * N + col] = (OutT)acc[i][j][r];
      }
}

// ---------------------------------------------------------------------------
// Causal flash attention, one wave per 16-query block, Dh = 64.
// Q,K: [B*S, D] bf16 ; Vt: [B,H,64,S] bf16 ; Ctx: [B*S, D] bf16
// ---------------------------------------------------------------------------
__global__ __launch_bounds__(256) void attn_causal(
    const __bf16* __restrict__ Q, const __bf16* __restrict__ Km,
    const __bf16* __restrict__ Vt, __bf16* __restrict__ Ctx,
    int S, int D, int H) {
  const int bh   = blockIdx.y;
  const int b    = bh / H;
  const int h    = bh % H;
  const int wave = threadIdx.x >> 5;
  const int lane = threadIdx.x & 31;
  const int qb   = blockIdx.x * 8 + wave;   // 16-row query block index
  const int q0   = qb * 16;

  __shared__ __align__(16) __bf16 Ps[8][16][32];  // per-wave P tile

  const size_t rowbase = (size_t)b * S;
  const __bf16* Qb = Q  + (rowbase + q0) * D + h * 64;
  const __bf16* Kb = Km + rowbase * D + h * 64;
  const __bf16* Vb = Vt + (size_t)bh * 64 * S;    // [64][S]

  const v16bf aq0 = load_frag_a(Qb,      D, lane);  // d = 0..31
  const v16bf aq1 = load_frag_a(Qb + 32, D, lane);  // d = 32..63

  v8f o0 = {}, o1 = {}, o2 = {}, o3 = {};
  float rmax[8], rsum[8];
#pragma unroll
  for (int r = 0; r < 8; ++r) { rmax[r] = -INFINITY; rsum[r] = 0.0f; }

  const float scale = 0.125f;  // 1/sqrt(64)
  const int mofs = (lane >> 4) ? 8 : 0;
  const int nlan = lane & 15;
  const int qhi  = q0 + 15;

  for (int kc = 0; kc <= qhi; kc += 32) {
    float p[2][8];
    // ---- scores: two 16x16 tiles over 32 keys --------------------------
#pragma unroll
    for (int nt = 0; nt < 2; ++nt) {
      const int kb = kc + nt * 16;
      v8f sc = {};
      sc = wmma_bf16(aq0, load_frag_b(Kb + (size_t)kb * D,      D, lane), sc);
      sc = wmma_bf16(aq1, load_frag_b(Kb + (size_t)kb * D + 32, D, lane), sc);
      const int key = kb + nlan;
#pragma unroll
      for (int r = 0; r < 8; ++r) {
        float v = sc[r] * scale;
        p[nt][r] = (key <= q0 + r + mofs) ? v : -INFINITY;
      }
    }
    // ---- online softmax: row reductions across the 16-lane N groups ----
    float tmax[8];
#pragma unroll
    for (int r = 0; r < 8; ++r) tmax[r] = fmaxf(p[0][r], p[1][r]);
#pragma unroll
    for (int off = 1; off < 16; off <<= 1)
#pragma unroll
      for (int r = 0; r < 8; ++r)
        tmax[r] = fmaxf(tmax[r], __shfl_xor(tmax[r], off, 32));

    float corr[8];
#pragma unroll
    for (int r = 0; r < 8; ++r) {
      float nm = fmaxf(rmax[r], tmax[r]);
      corr[r] = __expf(rmax[r] - nm);
      rmax[r] = nm;
    }
    float psum[8];
#pragma unroll
    for (int r = 0; r < 8; ++r) {
      p[0][r] = __expf(p[0][r] - rmax[r]);
      p[1][r] = __expf(p[1][r] - rmax[r]);
      psum[r] = p[0][r] + p[1][r];
    }
#pragma unroll
    for (int off = 1; off < 16; off <<= 1)
#pragma unroll
      for (int r = 0; r < 8; ++r)
        psum[r] += __shfl_xor(psum[r], off, 32);
#pragma unroll
    for (int r = 0; r < 8; ++r) rsum[r] = rsum[r] * corr[r] + psum[r];
#pragma unroll
    for (int r = 0; r < 8; ++r) {
      o0[r] *= corr[r]; o1[r] *= corr[r]; o2[r] *= corr[r]; o3[r] *= corr[r];
    }
    // ---- transpose P (C-layout -> A-layout) through per-wave LDS -------
#pragma unroll
    for (int nt = 0; nt < 2; ++nt)
#pragma unroll
      for (int r = 0; r < 8; ++r)
        Ps[wave][r + mofs][nt * 16 + nlan] = (__bf16)p[nt][r];
    const v16bf ap = load_frag_a(&Ps[wave][0][0], 32, lane);
    // ---- context accumulate: o += P * V --------------------------------
    o0 = wmma_bf16(ap, load_frag_b(Vb + (size_t)0  * S + kc, S, lane), o0);
    o1 = wmma_bf16(ap, load_frag_b(Vb + (size_t)16 * S + kc, S, lane), o1);
    o2 = wmma_bf16(ap, load_frag_b(Vb + (size_t)32 * S + kc, S, lane), o2);
    o3 = wmma_bf16(ap, load_frag_b(Vb + (size_t)48 * S + kc, S, lane), o3);
  }

  // ---- normalize and write context (bf16) ------------------------------
  __bf16* crow = Ctx + (rowbase + q0) * D + h * 64;
#pragma unroll
  for (int r = 0; r < 8; ++r) {
    float inv = 1.0f / rsum[r];
    int row = r + mofs;
    crow[(size_t)row * D +  0 + nlan] = (__bf16)(o0[r] * inv);
    crow[(size_t)row * D + 16 + nlan] = (__bf16)(o1[r] * inv);
    crow[(size_t)row * D + 32 + nlan] = (__bf16)(o2[r] * inv);
    crow[(size_t)row * D + 48 + nlan] = (__bf16)(o3[r] * inv);
  }
}

// ---------------------------------------------------------------------------
// Host launch
// ---------------------------------------------------------------------------
extern "C" void kernel_launch(void* const* d_in, const int* in_sizes, int n_in,
                              void* d_out, int out_size, void* d_ws, size_t ws_size,
                              hipStream_t stream) {
  (void)in_sizes; (void)n_in; (void)out_size; (void)ws_size;
  const int B = 2, S = 2048, D = 768, H = 12;
  const int M = B * S;

  const float* x  = (const float*)d_in[0];
  const float* Wq = (const float*)d_in[1];
  const float* Wk = (const float*)d_in[2];
  const float* Wv = (const float*)d_in[3];
  const float* Wo = (const float*)d_in[4];
  float* out = (float*)d_out;

  char* base = (char*)d_ws;
  size_t off = 0;
  auto alloc = [&](size_t bytes) -> void* {
    void* p = base + off;
    off += (bytes + 255) & ~(size_t)255;
    return p;
  };
  const size_t nx = (size_t)M * D;   // 3,145,728
  const size_t nw = (size_t)D * D;   //   589,824

  __bf16* xb  = (__bf16*)alloc(nx * 2);
  __bf16* wqb = (__bf16*)alloc(nw * 2);
  __bf16* wkb = (__bf16*)alloc(nw * 2);
  __bf16* wvb = (__bf16*)alloc(nw * 2);
  __bf16* wob = (__bf16*)alloc(nw * 2);
  __bf16* Qb  = (__bf16*)alloc(nx * 2);
  __bf16* Kb  = (__bf16*)alloc(nx * 2);
  __bf16* Vb  = (__bf16*)alloc(nx * 2);
  __bf16* Vtb = (__bf16*)alloc(nx * 2);
  __bf16* Ctx = (__bf16*)alloc(nx * 2);

  const int T = 256;
  f32_to_bf16<<<(unsigned)((nx + T - 1) / T), T, 0, stream>>>(x,  xb,  (int)nx);
  f32_to_bf16<<<(unsigned)((nw + T - 1) / T), T, 0, stream>>>(Wq, wqb, (int)nw);
  f32_to_bf16<<<(unsigned)((nw + T - 1) / T), T, 0, stream>>>(Wk, wkb, (int)nw);
  f32_to_bf16<<<(unsigned)((nw + T - 1) / T), T, 0, stream>>>(Wv, wvb, (int)nw);
  f32_to_bf16<<<(unsigned)((nw + T - 1) / T), T, 0, stream>>>(Wo, wob, (int)nw);

  dim3 gg(D / BN, M / BM);  // (6, 32)
  gemm_bf16_wmma<__bf16><<<gg, 256, 0, stream>>>(xb, wqb, Qb, M, D, D);
  gemm_bf16_wmma<__bf16><<<gg, 256, 0, stream>>>(xb, wkb, Kb, M, D, D);
  gemm_bf16_wmma<__bf16><<<gg, 256, 0, stream>>>(xb, wvb, Vb, M, D, D);

  transpose_v<<<(unsigned)((nx + T - 1) / T), T, 0, stream>>>(Vb, Vtb, B, S, D, H);

  attn_causal<<<dim3(S / 16 / 8, B * H), 256, 0, stream>>>(Qb, Kb, Vtb, Ctx, S, D, H);

  gemm_bf16_wmma<float><<<gg, 256, 0, stream>>>(Ctx, wob, out, M, D, D);
}
